// GRUODEBayes_85968065397115
// MI455X (gfx1250) — matrix operand order repaired and checked by reading, not compiled, not run on hardware
//
#include <hip/hip_runtime.h>
#include <hip/hip_bf16.h>

// ---------------------------------------------------------------------------
// GRU-ODE-Bayes for MI455X (gfx1250, wave32, WMMA).
// All GEMMs use v_wmma_f32_16x16x32_bf16 (bf16 in, f32 accumulate).
// ---------------------------------------------------------------------------

typedef __attribute__((ext_vector_type(16))) __bf16 v16bf;
typedef __attribute__((ext_vector_type(8)))  float  v8f;

union FragBF { uint4 u4[2]; v16bf v; };

// sizes
#define BB   256
#define TT   200
#define DD   32
#define HH   512
#define PHH  256
#define PREPN 16
#define DTSTEP 0.05f

__device__ __forceinline__ unsigned short f2bf(float f) {
  unsigned u = __builtin_bit_cast(unsigned, f);
  u += 0x7fffu + ((u >> 16) & 1u);           // round-to-nearest-even
  return (unsigned short)(u >> 16);
}

__device__ __forceinline__ float sigmoidf_(float x) {
  return 1.0f / (1.0f + __expf(-x));
}

// A operand: 16x32 bf16 tile of row-major A[*, lda]. lanes 0-15: rows m0..m0+15
// with K = k0..k0+15 contiguous; lanes 16-31: same rows, K = k0+16..k0+31.
__device__ __forceinline__ v16bf load_a_frag(const unsigned short* A, int lda,
                                             int row0, int k0) {
  int lane = threadIdx.x & 31;
  int row  = row0 + (lane & 15);
  int kb   = k0 + ((lane & 16) ? 16 : 0);
  const unsigned short* p = A + (size_t)row * lda + kb;
  FragBF f;
  f.u4[0] = *reinterpret_cast<const uint4*>(p);       // K kb..kb+7
  f.u4[1] = *reinterpret_cast<const uint4*>(p + 8);   // K kb+8..kb+15
  return f.v;
}

// B operand: 32x16 bf16 (K x N) taken from row-major weights W[N, ldw]
// (computing A @ W^T). lane n<16 -> column n0+n, K = k0..k0+15 contiguous;
// lanes 16-31 -> K = k0+16..k0+31.
__device__ __forceinline__ v16bf load_b_frag(const unsigned short* W, int ldw,
                                             int n0, int k0) {
  int lane = threadIdx.x & 31;
  int n    = n0 + (lane & 15);
  int kb   = k0 + ((lane & 16) ? 16 : 0);
  const unsigned short* p = W + (size_t)n * ldw + kb;
  FragBF f;
  f.u4[0] = *reinterpret_cast<const uint4*>(p);      // K kb..kb+7
  f.u4[1] = *reinterpret_cast<const uint4*>(p + 8);  // K kb+8..kb+15
  return f.v;
}

__device__ __forceinline__ v8f wmma_bf16(v16bf a, v16bf b, v8f c) {
  return __builtin_amdgcn_wmma_f32_16x16x32_bf16(false, a, false, b,
                                                 (short)0, c, false, false);
}

// ---------------------------------------------------------------------------
// Setup kernels
// ---------------------------------------------------------------------------
__global__ __launch_bounds__(256) void k_cvt(const float* __restrict__ src,
                                             unsigned short* __restrict__ dst,
                                             int n) {
  int i = blockIdx.x * 256 + threadIdx.x;
  if (i < n) dst[i] = f2bf(src[i]);
}

// h0 stage 1: tmp[b,c] = relu(cov @ Wc1^T + bc1), [256,128]
__global__ __launch_bounds__(256) void k_cov1(const float* __restrict__ cov,
                                              const float* __restrict__ Wc1,
                                              const float* __restrict__ bc1,
                                              float* __restrict__ tmp) {
  int i = blockIdx.x * 256 + threadIdx.x;  // 32768
  int b = i >> 7, c = i & 127;
  float s = bc1[c];
  #pragma unroll
  for (int k = 0; k < 32; ++k) s += cov[b * 32 + k] * Wc1[c * 32 + k];
  tmp[i] = fmaxf(s, 0.0f);
}

// h0 stage 2: h = tanh(tmp @ Wc2^T + bc2); write f32 + bf16 (h half of hx)
__global__ __launch_bounds__(256) void k_cov2(const float* __restrict__ tmp,
                                              const float* __restrict__ Wc2,
                                              const float* __restrict__ bc2,
                                              float* __restrict__ h32,
                                              unsigned short* __restrict__ hx) {
  int i = blockIdx.x * 256 + threadIdx.x;  // 131072
  int b = i >> 9, c = i & 511;
  float s = bc2[c];
  for (int k = 0; k < 128; ++k) s += tmp[b * 128 + k] * Wc2[c * 128 + k];
  float h = tanhf(s);
  h32[i] = h;
  hx[b * 1024 + 512 + c] = f2bf(h);
}

// obs[t,b] = any(M[t,b,:] > 0)
__global__ __launch_bounds__(256) void k_obs(const float* __restrict__ M,
                                             int* __restrict__ obs) {
  int i = blockIdx.x * 256 + threadIdx.x;  // 51200
  float s = 0.f;
  #pragma unroll
  for (int d = 0; d < DD; ++d) s += M[(size_t)i * DD + d];
  obs[i] = (s > 0.0f) ? 1 : 0;
}

// ---------------------------------------------------------------------------
// Per-step kernel 1: r,z = sigmoid(h @ [Whr|Whz]^T); emit rh (bf16), z (f32)
// 512 waves: 16 m-tiles x 32 j-tiles, 2 accumulators (r and z share col j).
// ---------------------------------------------------------------------------
__global__ __launch_bounds__(256) void k_rz(const unsigned short* __restrict__ hx,
                                            const unsigned short* __restrict__ wrz,
                                            const float* __restrict__ h32,
                                            float* __restrict__ z32,
                                            unsigned short* __restrict__ rh) {
  int wid = blockIdx.x * 8 + (threadIdx.x >> 5);
  int mt = wid >> 5, jt = wid & 31;
  v8f ar = {}, az = {};
  #pragma unroll
  for (int k0 = 0; k0 < HH; k0 += 32) {
    v16bf a  = load_a_frag(hx, 1024, mt * 16, 512 + k0);
    v16bf br = load_b_frag(wrz, 512, jt * 16, k0);
    v16bf bz = load_b_frag(wrz, 512, 512 + jt * 16, k0);
    ar = wmma_bf16(a, br, ar);
    az = wmma_bf16(a, bz, az);
  }
  int lane = threadIdx.x & 31;
  int col = jt * 16 + (lane & 15);
  int rbase = mt * 16 + ((lane & 16) ? 8 : 0);
  #pragma unroll
  for (int v = 0; v < 8; ++v) {
    int idx = (rbase + v) * HH + col;
    float r = sigmoidf_(ar[v]);
    float z = sigmoidf_(az[v]);
    z32[idx] = z;
    rh[idx] = f2bf(r * h32[idx]);
  }
}

// ---------------------------------------------------------------------------
// Per-step kernel 2: u = tanh(rh @ Whh^T); h = h + DT*(1-z)*(u-h) (Euler)
// ---------------------------------------------------------------------------
__global__ __launch_bounds__(256) void k_ode(const unsigned short* __restrict__ rh,
                                             const unsigned short* __restrict__ whh,
                                             const float* __restrict__ z32,
                                             float* __restrict__ h32,
                                             unsigned short* __restrict__ hx) {
  int wid = blockIdx.x * 8 + (threadIdx.x >> 5);
  int mt = wid >> 5, jt = wid & 31;
  v8f au = {};
  #pragma unroll
  for (int k0 = 0; k0 < HH; k0 += 32) {
    v16bf a = load_a_frag(rh, 512, mt * 16, k0);
    v16bf b = load_b_frag(whh, 512, jt * 16, k0);
    au = wmma_bf16(a, b, au);
  }
  int lane = threadIdx.x & 31;
  int col = jt * 16 + (lane & 15);
  int rbase = mt * 16 + ((lane & 16) ? 8 : 0);
  #pragma unroll
  for (int v = 0; v < 8; ++v) {
    int row = rbase + v;
    int idx = row * HH + col;
    float u = tanhf(au[v]);
    float hold = h32[idx];
    float z = z32[idx];
    float hn = hold + DTSTEP * (1.0f - z) * (u - hold);
    h32[idx] = hn;
    hx[row * 1024 + 512 + col] = f2bf(hn);
  }
}

// ---------------------------------------------------------------------------
// Per-step kernel 3: q = relu(h @ Wp1^T + bp1) -> bf16 [256,256]
// ---------------------------------------------------------------------------
__global__ __launch_bounds__(256) void k_p1(const unsigned short* __restrict__ hx,
                                            const unsigned short* __restrict__ wp1,
                                            const float* __restrict__ bp1,
                                            unsigned short* __restrict__ qb) {
  int wid = blockIdx.x * 8 + (threadIdx.x >> 5);
  int mt = wid >> 4, nt = wid & 15;
  v8f acc = {};
  #pragma unroll
  for (int k0 = 0; k0 < HH; k0 += 32) {
    v16bf a = load_a_frag(hx, 1024, mt * 16, 512 + k0);
    v16bf b = load_b_frag(wp1, 512, nt * 16, k0);
    acc = wmma_bf16(a, b, acc);
  }
  int lane = threadIdx.x & 31;
  int col = nt * 16 + (lane & 15);
  int rbase = mt * 16 + ((lane & 16) ? 8 : 0);
  float bias = bp1[col];
  #pragma unroll
  for (int v = 0; v < 8; ++v) {
    float q = fmaxf(acc[v] + bias, 0.0f);
    qb[(rbase + v) * PHH + col] = f2bf(q);
  }
}

// ---------------------------------------------------------------------------
// Per-step kernel 4: p = q @ Wp2^T + bp2 -> mean/logvar/err + per-feature
// prep matvec [4x16], relu, mask -> gi bf16 into cols 0..511 of hx.
// A wave handles n-tiles (t2, t2+2) so the same lane owns mean(d) & logvar(d).
// ---------------------------------------------------------------------------
__global__ __launch_bounds__(256) void k_p2_prep(
    const unsigned short* __restrict__ qb, const unsigned short* __restrict__ wp2,
    const float* __restrict__ bp2, const float* __restrict__ X,
    const float* __restrict__ M, const float* __restrict__ wprep,
    const float* __restrict__ bprep, unsigned short* __restrict__ hx, int t) {
  int wid = blockIdx.x * 8 + (threadIdx.x >> 5);  // 32 waves
  int mt = wid >> 1, t2 = wid & 1;
  v8f a0 = {}, a1 = {};
  #pragma unroll
  for (int k0 = 0; k0 < PHH; k0 += 32) {
    v16bf a  = load_a_frag(qb, PHH, mt * 16, k0);
    v16bf b0 = load_b_frag(wp2, PHH, t2 * 16, k0);
    v16bf b1 = load_b_frag(wp2, PHH, 32 + t2 * 16, k0);
    a0 = wmma_bf16(a, b0, a0);
    a1 = wmma_bf16(a, b1, a1);
  }
  int lane = threadIdx.x & 31;
  int d = t2 * 16 + (lane & 15);
  int rbase = mt * 16 + ((lane & 16) ? 8 : 0);
  float wp[4][PREPN], bp[PREPN];
  #pragma unroll
  for (int f = 0; f < 4; ++f)
    #pragma unroll
    for (int p = 0; p < PREPN; ++p) wp[f][p] = wprep[d * 64 + f * PREPN + p];
  #pragma unroll
  for (int p = 0; p < PREPN; ++p) bp[p] = bprep[d * PREPN + p];
  float bm = bp2[d], bl = bp2[32 + d];
  #pragma unroll
  for (int v = 0; v < 8; ++v) {
    int row = rbase + v;
    float mean   = a0[v] + bm;
    float logvar = a1[v] + bl;
    size_t xi = ((size_t)t * BB + row) * DD + d;
    float x = X[xi];
    float m = M[xi];
    float err = (x - mean) * __expf(-0.5f * logvar);
    #pragma unroll
    for (int p = 0; p < PREPN; ++p) {
      float g = bp[p] + x * wp[0][p] + mean * wp[1][p] + logvar * wp[2][p] +
                err * wp[3][p];
      g = fmaxf(g, 0.0f) * m;
      hx[row * 1024 + d * PREPN + p] = f2bf(g);
    }
  }
}

// ---------------------------------------------------------------------------
// Per-step kernel 5: GRU cell. Four accumulators per wave (r, z, xn, hn):
//   r/z accumulate over K=1024 ([gi|h]); xn over gi only; hn over h only.
// Epilogue fuses the gate math and the obs scatter; writes new h into hx_out
// (double-buffered to avoid intra-kernel WAR on the h operand).
// ---------------------------------------------------------------------------
__global__ __launch_bounds__(256) void k_gates(
    const unsigned short* __restrict__ hx, const unsigned short* __restrict__ wih,
    const unsigned short* __restrict__ whh, const float* __restrict__ bih,
    const float* __restrict__ bhh, const int* __restrict__ obs,
    float* __restrict__ h32, unsigned short* __restrict__ hx_out, int t) {
  int wid = blockIdx.x * 8 + (threadIdx.x >> 5);
  int mt = wid >> 5, jt = wid & 31;
  v8f ar = {}, az = {}, axn = {}, ahn = {};
  #pragma unroll
  for (int k0 = 0; k0 < HH; k0 += 32) {  // gi half
    v16bf a = load_a_frag(hx, 1024, mt * 16, k0);
    ar  = wmma_bf16(a, load_b_frag(wih, 512, jt * 16, k0), ar);
    az  = wmma_bf16(a, load_b_frag(wih, 512, 512 + jt * 16, k0), az);
    axn = wmma_bf16(a, load_b_frag(wih, 512, 1024 + jt * 16, k0), axn);
  }
  #pragma unroll
  for (int k0 = 0; k0 < HH; k0 += 32) {  // h half
    v16bf a = load_a_frag(hx, 1024, mt * 16, 512 + k0);
    ar  = wmma_bf16(a, load_b_frag(whh, 512, jt * 16, k0), ar);
    az  = wmma_bf16(a, load_b_frag(whh, 512, 512 + jt * 16, k0), az);
    ahn = wmma_bf16(a, load_b_frag(whh, 512, 1024 + jt * 16, k0), ahn);
  }
  int lane = threadIdx.x & 31;
  int col = jt * 16 + (lane & 15);
  int rbase = mt * 16 + ((lane & 16) ? 8 : 0);
  float br  = bih[col] + bhh[col];
  float bz  = bih[512 + col] + bhh[512 + col];
  float bxn = bih[1024 + col];
  float bhn = bhh[1024 + col];
  #pragma unroll
  for (int v = 0; v < 8; ++v) {
    int row = rbase + v;
    int idx = row * HH + col;
    float rg = sigmoidf_(ar[v] + br);
    float zg = sigmoidf_(az[v] + bz);
    float ng = tanhf(axn[v] + bxn + rg * (ahn[v] + bhn));
    float hold = h32[idx];
    float hn = obs[t * BB + row] ? (1.0f - zg) * ng + zg * hold : hold;
    h32[idx] = hn;
    hx_out[row * 1024 + 512 + col] = f2bf(hn);
  }
}

// ---------------------------------------------------------------------------
// Per-step kernel 6: preds = relu(h @ Wclf1^T + b) @ Wclf2^T + b.
// One wave per m-tile carries all 4 n-tiles; 64->1 reduction via shfl_xor.
// ---------------------------------------------------------------------------
__global__ __launch_bounds__(256) void k_clf(const unsigned short* __restrict__ hx,
                                             const unsigned short* __restrict__ wclf1,
                                             const float* __restrict__ bclf1,
                                             const float* __restrict__ Wclf2,
                                             const float* __restrict__ bclf2,
                                             float* __restrict__ out, int t) {
  int wid = blockIdx.x * 8 + (threadIdx.x >> 5);  // 16 waves
  int mt = wid;
  v8f acc[4] = {{}, {}, {}, {}};
  #pragma unroll
  for (int k0 = 0; k0 < HH; k0 += 32) {
    v16bf a = load_a_frag(hx, 1024, mt * 16, 512 + k0);
    #pragma unroll
    for (int nt = 0; nt < 4; ++nt)
      acc[nt] = wmma_bf16(a, load_b_frag(wclf1, 512, nt * 16, k0), acc[nt]);
  }
  int lane = threadIdx.x & 31;
  int ci = lane & 15;
  int rbase = mt * 16 + ((lane & 16) ? 8 : 0);
  float w2[4], b1[4];
  #pragma unroll
  for (int nt = 0; nt < 4; ++nt) {
    w2[nt] = Wclf2[nt * 16 + ci];
    b1[nt] = bclf1[nt * 16 + ci];
  }
  float bout = bclf2[0];
  #pragma unroll
  for (int v = 0; v < 8; ++v) {
    float s = 0.f;
    #pragma unroll
    for (int nt = 0; nt < 4; ++nt)
      s += fmaxf(acc[nt][v] + b1[nt], 0.0f) * w2[nt];
    s += __shfl_xor(s, 1, 32);
    s += __shfl_xor(s, 2, 32);
    s += __shfl_xor(s, 4, 32);
    s += __shfl_xor(s, 8, 32);
    if (ci == 0) out[t * BB + rbase + v] = s + bout;
  }
}

// ---------------------------------------------------------------------------
extern "C" void kernel_launch(void* const* d_in, const int* in_sizes, int n_in,
                              void* d_out, int out_size, void* d_ws, size_t ws_size,
                              hipStream_t stream) {
  (void)in_sizes; (void)n_in; (void)out_size; (void)ws_size;
  const float* cov   = (const float*)d_in[0];
  const float* X     = (const float*)d_in[1];
  const float* M     = (const float*)d_in[2];
  const float* Wc1   = (const float*)d_in[3];
  const float* bc1   = (const float*)d_in[4];
  const float* Wc2   = (const float*)d_in[5];
  const float* bc2   = (const float*)d_in[6];
  const float* Whr   = (const float*)d_in[7];
  const float* Whz   = (const float*)d_in[8];
  const float* Whh   = (const float*)d_in[9];
  const float* Wp1   = (const float*)d_in[10];
  const float* bp1   = (const float*)d_in[11];
  const float* Wp2   = (const float*)d_in[12];
  const float* bp2   = (const float*)d_in[13];
  const float* wprep = (const float*)d_in[14];
  const float* bprep = (const float*)d_in[15];
  const float* Wih   = (const float*)d_in[16];
  const float* Whhg  = (const float*)d_in[17];
  const float* bih   = (const float*)d_in[18];
  const float* bhh   = (const float*)d_in[19];
  const float* Wclf1 = (const float*)d_in[20];
  const float* bclf1 = (const float*)d_in[21];
  const float* Wclf2 = (const float*)d_in[22];
  const float* bclf2 = (const float*)d_in[23];
  float* out = (float*)d_out;

  char* ws = (char*)d_ws;
  size_t off = 0;
  auto carve = [&](size_t bytes) -> char* {
    char* p = ws + off;
    off = (off + bytes + 255) & ~(size_t)255;
    return p;
  };
  unsigned short* wrz_bf   = (unsigned short*)carve((size_t)1024 * 512 * 2);
  unsigned short* whh_bf   = (unsigned short*)carve((size_t)512 * 512 * 2);
  unsigned short* wp1_bf   = (unsigned short*)carve((size_t)256 * 512 * 2);
  unsigned short* wp2_bf   = (unsigned short*)carve((size_t)64 * 256 * 2);
  unsigned short* wih_bf   = (unsigned short*)carve((size_t)1536 * 512 * 2);
  unsigned short* whhg_bf  = (unsigned short*)carve((size_t)1536 * 512 * 2);
  unsigned short* wclf1_bf = (unsigned short*)carve((size_t)64 * 512 * 2);
  float*          h32      = (float*)carve((size_t)BB * HH * 4);
  unsigned short* hxA      = (unsigned short*)carve((size_t)BB * 1024 * 2);
  unsigned short* hxB      = (unsigned short*)carve((size_t)BB * 1024 * 2);
  float*          z32      = (float*)carve((size_t)BB * HH * 4);
  unsigned short* rh_bf    = (unsigned short*)carve((size_t)BB * HH * 2);
  unsigned short* q_bf     = (unsigned short*)carve((size_t)BB * PHH * 2);
  float*          tmp_cov  = (float*)carve((size_t)BB * 128 * 4);
  int*            obs      = (int*)carve((size_t)TT * BB * 4);

  // --- setup: weight conversion (f32 -> bf16), h0, obs flags ---
  auto cvt = [&](const float* s, unsigned short* d, int n) {
    k_cvt<<<(n + 255) / 256, 256, 0, stream>>>(s, d, n);
  };
  cvt(Whr, wrz_bf, 512 * 512);
  cvt(Whz, wrz_bf + 512 * 512, 512 * 512);
  cvt(Whh, whh_bf, 512 * 512);
  cvt(Wp1, wp1_bf, 256 * 512);
  cvt(Wp2, wp2_bf, 64 * 256);
  cvt(Wih, wih_bf, 1536 * 512);
  cvt(Whhg, whhg_bf, 1536 * 512);
  cvt(Wclf1, wclf1_bf, 64 * 512);

  k_cov1<<<128, 256, 0, stream>>>(cov, Wc1, bc1, tmp_cov);
  k_cov2<<<512, 256, 0, stream>>>(tmp_cov, Wc2, bc2, h32, hxA);
  k_obs<<<200, 256, 0, stream>>>(M, obs);

  // --- sequential scan over T steps ---
  for (int t = 0; t < TT; ++t) {
    unsigned short* cur = (t & 1) ? hxB : hxA;
    unsigned short* nxt = (t & 1) ? hxA : hxB;
    k_rz<<<64, 256, 0, stream>>>(cur, wrz_bf, h32, z32, rh_bf);
    k_ode<<<64, 256, 0, stream>>>(rh_bf, whh_bf, z32, h32, cur);
    k_p1<<<32, 256, 0, stream>>>(cur, wp1_bf, bp1, q_bf);
    k_p2_prep<<<4, 256, 0, stream>>>(q_bf, wp2_bf, bp2, X, M, wprep, bprep,
                                     cur, t);
    k_gates<<<64, 256, 0, stream>>>(cur, wih_bf, whhg_bf, bih, bhh, obs, h32,
                                    nxt, t);
    k_clf<<<2, 256, 0, stream>>>(nxt, wclf1_bf, bclf1, Wclf2, bclf2, out, t);
  }
}
